// MpnnGGNN_45543833207155
// MI455X (gfx1250) — matrix operand order repaired and checked by reading, not compiled, not run on hardware
//
#include <hip/hip_runtime.h>
#include <math.h>

// ---------------- types ----------------
typedef __bf16 bf16;
typedef __attribute__((ext_vector_type(16))) __bf16 v16bf;
typedef __attribute__((ext_vector_type(8)))  __bf16 v8bf;
typedef __attribute__((ext_vector_type(8)))  float  v8f;

#define SHUF16(lo, hi) __builtin_shufflevector(lo, hi, 0,1,2,3,4,5,6,7,8,9,10,11,12,13,14,15)

// ---------------- problem constants ----------------
#define BB   64
#define NNODE 128
#define FIN  128
#define HH   512
#define LL   4
#define ROWS (BB*NNODE)          // 8192
#define NSTEPS 4

__device__ __forceinline__ bf16 f2bf(float x) {
  unsigned u = __builtin_bit_cast(unsigned, x);
  unsigned r = u + 0x7FFFu + ((u >> 16) & 1u);
  unsigned short h = (unsigned short)(r >> 16);
  return __builtin_bit_cast(bf16, h);
}
__device__ __forceinline__ float sigmoidf_(float x) { return 1.0f / (1.0f + expf(-x)); }

// ======================================================================
// Generic bf16 WMMA GEMM: C[M x Nreal] = act( A[M x K] * Bsw + bias )
// A is row-major with an optional K-split across two buffers (concat).
// Bsw is pre-swizzled: element i of lane L of tile (kt,nt) =
//   B[kt*32 + (L>>4)*16 + i][nt*16 + (L&15)], stored 16-contiguous per lane.
// Block = 128 threads = 4 waves, each wave owns one 16-col tile.
// ======================================================================
__global__ void gemm_ws(const bf16* __restrict__ A1, int lda1,
                        const bf16* __restrict__ A2, int lda2, int ksplit,
                        const bf16* __restrict__ Bsw, int K, int Ncols, int Nreal,
                        const float* __restrict__ bias,
                        int relu, int outbf,
                        void* __restrict__ C, int ldc)
{
  int lane  = threadIdx.x & 31;
  int wv    = threadIdx.x >> 5;
  int ntile = blockIdx.x * 4 + wv;
  if (ntile * 16 >= Ncols) return;            // wave-uniform: EXEC stays full
  int mtile = blockIdx.y;
  int am = mtile * 16 + (lane & 15);
  int kb = (lane >> 4) * 16;
  int Ktiles = K >> 5;
  v8f acc = {0.f,0.f,0.f,0.f,0.f,0.f,0.f,0.f};
  for (int k0 = 0; k0 < K; k0 += 32) {
    const bf16* ap = (k0 < ksplit)
        ? (A1 + (size_t)am * lda1 + k0 + kb)
        : (A2 + (size_t)am * lda2 + (k0 - ksplit) + kb);
    v8bf alo = *(const v8bf*)(ap);
    v8bf ahi = *(const v8bf*)(ap + 8);
    v16bf a = SHUF16(alo, ahi);
    const v8bf* bp = (const v8bf*)(Bsw + ((((size_t)ntile * Ktiles + (k0 >> 5)) * 32 + lane) << 4));
    v16bf b = SHUF16(bp[0], bp[1]);
    acc = __builtin_amdgcn_wmma_f32_16x16x32_bf16(false, a, false, b, (short)0, acc, false, false);
  }
  int cn = ntile * 16 + (lane & 15);
  if (cn >= Nreal) return;
  float bv = bias ? bias[cn] : 0.0f;
  int rbase = mtile * 16 + ((lane >> 4) << 3);
  #pragma unroll
  for (int j = 0; j < 8; ++j) {
    float v = acc[j] + bv;
    if (relu) v = fmaxf(v, 0.0f);
    size_t o = (size_t)(rbase + j) * ldc + cn;
    if (outbf) ((bf16*)C)[o] = f2bf(v);
    else       ((float*)C)[o] = v;
  }
}

// ======================================================================
// Aggregation: S[b][v][l*512+h'] = sum_w adj[b,l][v][w] * h[b][w][h']
// adj row-major A-operand, hB swizzled B-operand (per batch, K=128,N=512).
// grid = (8, 8, B*L), block = 128 (4 waves).
// ======================================================================
__global__ void aggregate_ws(const bf16* __restrict__ adj,
                             const bf16* __restrict__ hB,
                             bf16* __restrict__ S)
{
  int lane  = threadIdx.x & 31;
  int wv    = threadIdx.x >> 5;
  int ntile = blockIdx.x * 4 + wv;            // 0..31 over 512 cols
  int mtile = blockIdx.y;                     // 0..7
  int z = blockIdx.z;                         // b*4 + l
  int b = z >> 2, l = z & 3;
  const bf16* A  = adj + (size_t)z * (NNODE*NNODE);
  const bf16* Bs = hB  + (size_t)b * (NNODE*HH);
  int am = mtile * 16 + (lane & 15);
  int kb = (lane >> 4) * 16;
  const int Ktiles = NNODE / 32;              // 4
  v8f acc = {0.f,0.f,0.f,0.f,0.f,0.f,0.f,0.f};
  #pragma unroll
  for (int k0 = 0; k0 < NNODE; k0 += 32) {
    const bf16* ap = A + (size_t)am * NNODE + k0 + kb;
    v8bf alo = *(const v8bf*)(ap);
    v8bf ahi = *(const v8bf*)(ap + 8);
    v16bf a = SHUF16(alo, ahi);
    const v8bf* bp = (const v8bf*)(Bs + ((((size_t)ntile * Ktiles + (k0 >> 5)) * 32 + lane) << 4));
    v16bf bfr = SHUF16(bp[0], bp[1]);
    acc = __builtin_amdgcn_wmma_f32_16x16x32_bf16(false, a, false, bfr, (short)0, acc, false, false);
  }
  int cn = ntile * 16 + (lane & 15);
  int rbase = mtile * 16 + ((lane >> 4) << 3);
  bf16* Cp = S + (size_t)b * (NNODE*4*HH) + (size_t)l * HH;
  #pragma unroll
  for (int j = 0; j < 8; ++j)
    Cp[(size_t)(rbase + j) * (4*HH) + cn] = f2bf(acc[j]);
}

// ======================================================================
// Fused GRU step: one wave computes the 4 gate tiles (r, z, i_n, h_n) for
// 16 rows x 16 hidden units via Wcat [1024 x 2048] (zero blocks in n-cols),
// input A = [m | h_old]. Then applies GRU update in-register and writes
// h (f32 state), h bf16 row-major (next A operand), hB swizzled (next B op).
// grid = (32, 512), block = 32 (1 wave).
// ======================================================================
__global__ void gru_ws(const bf16* __restrict__ mbuf,    // [8192][512]
                       const bf16* __restrict__ hbf_in,  // [8192][512]
                       const bf16* __restrict__ Wcat,    // swizzled K=1024,N=2048
                       const float* __restrict__ bih, const float* __restrict__ bhh,
                       const float* __restrict__ nodemask,
                       float* __restrict__ hstate,
                       bf16* __restrict__ hbf_out,
                       bf16* __restrict__ hB)
{
  int lane  = threadIdx.x & 31;
  int ntile = blockIdx.x;                     // 0..31 (hidden col tiles)
  int mtile = blockIdx.y;                     // 0..511
  int am = mtile * 16 + (lane & 15);
  int kb = (lane >> 4) * 16;
  const int Ktiles = 32;                      // 1024/32
  v8f zer = {0.f,0.f,0.f,0.f,0.f,0.f,0.f,0.f};
  v8f acc[4] = {zer, zer, zer, zer};          // r, z, i_n, h_n
  for (int k0 = 0; k0 < 1024; k0 += 32) {
    const bf16* ap = (k0 < 512)
        ? (mbuf   + (size_t)am * HH + k0 + kb)
        : (hbf_in + (size_t)am * HH + (k0 - 512) + kb);
    v8bf alo = *(const v8bf*)(ap);
    v8bf ahi = *(const v8bf*)(ap + 8);
    v16bf a = SHUF16(alo, ahi);
    int kt = k0 >> 5;
    #pragma unroll
    for (int gidx = 0; gidx < 4; ++gidx) {
      int nt = ntile + gidx * 32;
      const v8bf* bp = (const v8bf*)(Wcat + ((((size_t)nt * Ktiles + kt) * 32 + lane) << 4));
      v16bf b = SHUF16(bp[0], bp[1]);
      acc[gidx] = __builtin_amdgcn_wmma_f32_16x16x32_bf16(false, a, false, b, (short)0, acc[gidx], false, false);
    }
  }
  int hcol = ntile * 16 + (lane & 15);
  float br  = bih[hcol]        + bhh[hcol];
  float bz  = bih[512 + hcol]  + bhh[512 + hcol];
  float bin = bih[1024 + hcol];
  float bhn = bhh[1024 + hcol];
  int rbase = mtile * 16 + ((lane >> 4) << 3);
  #pragma unroll
  for (int j = 0; j < 8; ++j) {
    int r = rbase + j;
    float rg = sigmoidf_(acc[0][j] + br);
    float zg = sigmoidf_(acc[1][j] + bz);
    float ng = tanhf(acc[2][j] + bin + rg * (acc[3][j] + bhn));
    float hold = hstate[(size_t)r * HH + hcol];
    float hn = ((1.0f - zg) * ng + zg * hold) * nodemask[r];
    hstate[(size_t)r * HH + hcol] = hn;
    bf16 hb = f2bf(hn);
    hbf_out[(size_t)r * HH + hcol] = hb;
    int b = r >> 7, w = r & 127;
    int kt2 = w >> 5, nt2 = hcol >> 4;
    int lane2 = (hcol & 15) + (((w >> 4) & 1) << 4);
    int j2 = w & 15;
    hB[(size_t)b * (NNODE*HH) + ((((size_t)nt2 * 4 + kt2) * 32 + lane2) << 4) + j2] = hb;
  }
}

// ======================================================================
// Packing / conversion kernels (run once per call)
// ======================================================================
__global__ void pack_adj(const float* __restrict__ g, const float* __restrict__ e,
                         bf16* __restrict__ adj)
{
  int idx = blockIdx.x * blockDim.x + threadIdx.x;      // over B*N*N
  if (idx >= BB*NNODE*NNODE) return;
  float gv = g[idx], ev = e[idx];
  int b  = idx / (NNODE*NNODE);
  int vw = idx % (NNODE*NNODE);
  #pragma unroll
  for (int l = 0; l < LL; ++l) {
    float val = (ev == (float)(l + 1)) ? gv : 0.0f;
    adj[(size_t)(b*LL + l) * (NNODE*NNODE) + vw] = f2bf(val);
  }
}

__global__ void pack_masks(const float* __restrict__ h_in,
                           float* __restrict__ nodemask, float* __restrict__ readmask)
{
  int r = blockIdx.x * blockDim.x + threadIdx.x;
  if (r >= ROWS) return;
  float sa = 0.f, ss = 0.f;
  for (int f = 0; f < FIN; ++f) { float x = h_in[(size_t)r*FIN + f]; sa += fabsf(x); ss += x; }
  nodemask[r] = (sa > 0.0f) ? 1.0f : 0.0f;
  readmask[r] = (ss != 0.0f) ? 1.0f : 0.0f;
}

__global__ void pack_h(const float* __restrict__ h_in,
                       float* __restrict__ hstate, bf16* __restrict__ hbf,
                       bf16* __restrict__ hB, bf16* __restrict__ h0b)
{
  size_t i = (size_t)blockIdx.x * blockDim.x + threadIdx.x;   // over ROWS*HH
  if (i >= (size_t)ROWS * HH) return;
  int r = (int)(i / HH), c = (int)(i % HH);
  float v = (c < FIN) ? h_in[(size_t)r*FIN + c] : 0.0f;
  hstate[i] = v;
  bf16 hb = f2bf(v);
  hbf[i] = hb;
  if (c < FIN) h0b[(size_t)r*FIN + c] = hb;
  int b = r >> 7, w = r & 127;
  int kt = w >> 5, nt = c >> 4;
  int lane2 = (c & 15) + (((w >> 4) & 1) << 4);
  int j2 = w & 15;
  hB[(size_t)b * (NNODE*HH) + ((((size_t)nt * 4 + kt) * 32 + lane2) << 4) + j2] = hb;
}

__global__ void pack_Abig(const float* __restrict__ A, bf16* __restrict__ AbigB)
{
  size_t i = (size_t)blockIdx.x * blockDim.x + threadIdx.x;   // K=2048 x N=512
  if (i >= 2048ull * 512ull) return;
  int k = (int)(i >> 9), n = (int)(i & 511);
  int l = k >> 9, hh = k & 511;
  float v = A[((size_t)l * HH + hh) * HH + n];
  int kt = k >> 5, nt = n >> 4;
  int lane2 = (n & 15) + (((k >> 4) & 1) << 4);
  int j2 = k & 15;
  AbigB[((((size_t)nt * 64 + kt) * 32 + lane2) << 4) + j2] = f2bf(v);
}

__global__ void pack_Wcat(const float* __restrict__ Wih, const float* __restrict__ Whh,
                          bf16* __restrict__ WcatB)
{
  size_t i = (size_t)blockIdx.x * blockDim.x + threadIdx.x;   // K=1024 x N=2048
  if (i >= 1024ull * 2048ull) return;
  int k = (int)(i >> 11), n = (int)(i & 2047);
  float v;
  if (n < 1024)       v = (k < 512) ? Wih[(size_t)n * HH + k] : Whh[(size_t)n * HH + (k - 512)];
  else if (n < 1536)  v = (k < 512) ? Wih[(size_t)n * HH + k] : 0.0f;                 // i_n
  else                v = (k >= 512) ? Whh[(size_t)(n - 512) * HH + (k - 512)] : 0.0f; // h_n
  int kt = k >> 5, nt = n >> 4;
  int lane2 = (n & 15) + (((k >> 4) & 1) << 4);
  int j2 = k & 15;
  WcatB[((((size_t)nt * 32 + kt) * 32 + lane2) << 4) + j2] = f2bf(v);
}

__global__ void pack_mlpw(const float* __restrict__ W, int nOut, int nIn,
                          int Kuse, int Npad, bf16* __restrict__ dst)
{
  int i = blockIdx.x * blockDim.x + threadIdx.x;
  if (i >= Kuse * Npad) return;
  int k = i / Npad, n = i % Npad;
  float v = (n < nOut) ? W[(size_t)n * nIn + k] : 0.0f;
  int Ktiles = Kuse >> 5;
  int kt = k >> 5, nt = n >> 4;
  int lane2 = (n & 15) + (((k >> 4) & 1) << 4);
  int j2 = k & 15;
  dst[((((size_t)nt * Ktiles + kt) * 32 + lane2) << 4) + j2] = f2bf(v);
}

// out[b,t] = sum_v sigmoid(gatepre) * valpre * readmask
__global__ void readout_reduce(const float* __restrict__ gatepre,
                               const float* __restrict__ valpre,
                               const float* __restrict__ readmask,
                               float* __restrict__ out)
{
  __shared__ float red[128][12];
  int b = blockIdx.x;
  int v = threadIdx.x;                        // 128
  int r = b * NNODE + v;
  float msk = readmask[r];
  #pragma unroll
  for (int t = 0; t < 12; ++t)
    red[v][t] = sigmoidf_(gatepre[(size_t)r * 16 + t]) * valpre[(size_t)r * 16 + t] * msk;
  __syncthreads();
  for (int s = 64; s > 0; s >>= 1) {
    if (v < s)
      for (int t = 0; t < 12; ++t) red[v][t] += red[v + s][t];
    __syncthreads();
  }
  if (v < 12) out[(size_t)b * 12 + v] = red[0][v];
}

// ======================================================================
extern "C" void kernel_launch(void* const* d_in, const int* in_sizes, int n_in,
                              void* d_out, int out_size, void* d_ws, size_t ws_size,
                              hipStream_t stream) {
  (void)in_sizes; (void)n_in; (void)out_size; (void)ws_size;
  const float* g    = (const float*)d_in[0];
  const float* h_in = (const float*)d_in[1];
  const float* e    = (const float*)d_in[2];
  const float* A    = (const float*)d_in[3];
  const float* Wih  = (const float*)d_in[4];
  const float* Whh  = (const float*)d_in[5];
  const float* bih  = (const float*)d_in[6];
  const float* bhh  = (const float*)d_in[7];
  const float* r1W[4] = {(const float*)d_in[8],  (const float*)d_in[12], (const float*)d_in[16], (const float*)d_in[20]};
  const float* r1b[4] = {(const float*)d_in[9],  (const float*)d_in[13], (const float*)d_in[17], (const float*)d_in[21]};
  const float* r2W[4] = {(const float*)d_in[10], (const float*)d_in[14], (const float*)d_in[18], (const float*)d_in[22]};
  const float* r2b[4] = {(const float*)d_in[11], (const float*)d_in[15], (const float*)d_in[19], (const float*)d_in[23]};
  float* out = (float*)d_out;

  // ---- workspace carve-up ----
  char* ws = (char*)d_ws; size_t off = 0;
  auto alloc = [&](size_t bytes) { char* p = ws + off; off += (bytes + 255) & ~(size_t)255; return p; };
  bf16* adj    = (bf16*)alloc((size_t)BB*LL*NNODE*NNODE*2);     // 8 MB
  float* hstate= (float*)alloc((size_t)ROWS*HH*4);              // 16 MB
  bf16* hbfA   = (bf16*)alloc((size_t)ROWS*HH*2);               // 8 MB
  bf16* hbfB   = (bf16*)alloc((size_t)ROWS*HH*2);               // 8 MB
  bf16* hB     = (bf16*)alloc((size_t)BB*NNODE*HH*2);           // 8 MB (swizzled)
  bf16* h0b    = (bf16*)alloc((size_t)ROWS*FIN*2);              // 2 MB
  bf16* S      = (bf16*)alloc((size_t)ROWS*4*HH*2);             // 32 MB
  bf16* mbuf   = (bf16*)alloc((size_t)ROWS*HH*2);               // 8 MB
  bf16* AbigB  = (bf16*)alloc(2048ull*512*2);                   // 2 MB
  bf16* WcatB  = (bf16*)alloc(1024ull*2048*2);                  // 4 MB
  bf16* r1w0s  = (bf16*)alloc(640ull*128*2);
  bf16* r1w1s  = (bf16*)alloc(128ull*256*2);
  bf16* r1w2s  = (bf16*)alloc(256ull*128*2);
  bf16* r1w3s  = (bf16*)alloc(128ull*16*2);
  bf16* r2w0s  = (bf16*)alloc(512ull*128*2);
  bf16* r2w1s  = (bf16*)alloc(128ull*256*2);
  bf16* r2w2s  = (bf16*)alloc(256ull*128*2);
  bf16* r2w3s  = (bf16*)alloc(128ull*16*2);
  bf16* act1   = (bf16*)alloc((size_t)ROWS*128*2);
  bf16* act2   = (bf16*)alloc((size_t)ROWS*256*2);
  bf16* act3   = (bf16*)alloc((size_t)ROWS*128*2);
  float* gatep = (float*)alloc((size_t)ROWS*16*4);
  float* valp  = (float*)alloc((size_t)ROWS*16*4);
  float* nmask = (float*)alloc((size_t)ROWS*4);
  float* rmask = (float*)alloc((size_t)ROWS*4);

  // ---- pack / convert ----
  pack_adj  <<<(BB*NNODE*NNODE + 255)/256, 256, 0, stream>>>(g, e, adj);
  pack_masks<<<(ROWS + 255)/256,           256, 0, stream>>>(h_in, nmask, rmask);
  pack_h    <<<(ROWS*HH + 255)/256,        256, 0, stream>>>(h_in, hstate, hbfA, hB, h0b);
  pack_Abig <<<(2048*512 + 255)/256,       256, 0, stream>>>(A, AbigB);
  pack_Wcat <<<(1024*2048 + 255)/256,      256, 0, stream>>>(Wih, Whh, WcatB);
  pack_mlpw <<<(640*128 + 255)/256, 256, 0, stream>>>(r1W[0], 128, 1024, 640, 128, r1w0s);
  pack_mlpw <<<(128*256 + 255)/256, 256, 0, stream>>>(r1W[1], 256, 128, 128, 256, r1w1s);
  pack_mlpw <<<(256*128 + 255)/256, 256, 0, stream>>>(r1W[2], 128, 256, 256, 128, r1w2s);
  pack_mlpw <<<(128*16  + 255)/256, 256, 0, stream>>>(r1W[3], 12, 128, 128, 16, r1w3s);
  pack_mlpw <<<(512*128 + 255)/256, 256, 0, stream>>>(r2W[0], 128, 512, 512, 128, r2w0s);
  pack_mlpw <<<(128*256 + 255)/256, 256, 0, stream>>>(r2W[1], 256, 128, 128, 256, r2w1s);
  pack_mlpw <<<(256*128 + 255)/256, 256, 0, stream>>>(r2W[2], 128, 256, 256, 128, r2w2s);
  pack_mlpw <<<(128*16  + 255)/256, 256, 0, stream>>>(r2W[3], 12, 128, 128, 16, r2w3s);

  // ---- message-passing steps ----
  bf16* hcur = hbfA; bf16* hnext = hbfB;
  for (int step = 0; step < NSTEPS; ++step) {
    // S[b] = Adj_l[b] @ h[b]  (per label, concatenated along K)
    aggregate_ws<<<dim3(8, 8, BB*LL), 128, 0, stream>>>(adj, hB, S);
    // m = S @ Abig   ([8192 x 2048] x [2048 x 512])
    gemm_ws<<<dim3(8, ROWS/16), 128, 0, stream>>>(
        S, 4*HH, S, 4*HH, 4*HH, AbigB, 4*HH, HH, HH,
        nullptr, 0, 1, mbuf, HH);
    // fused GRU: [m | h] @ Wcat -> gates -> h update
    gru_ws<<<dim3(32, ROWS/16), 32, 0, stream>>>(
        mbuf, hcur, WcatB, bih, bhh, nmask, hstate, hnext, hB);
    bf16* t = hcur; hcur = hnext; hnext = t;
  }

  // ---- readout: gate = sigmoid(MLP1([h,h0])), val = MLP2(h) ----
  // r1 layer 1: K=640 = [h(512) | h_in(128)], zero tail of h0 dropped
  gemm_ws<<<dim3(2, ROWS/16), 128, 0, stream>>>(hcur, HH, h0b, FIN, HH, r1w0s, 640, 128, 128,
                                                r1b[0], 1, 1, act1, 128);
  gemm_ws<<<dim3(4, ROWS/16), 128, 0, stream>>>(act1, 128, act1, 128, 128, r1w1s, 128, 256, 256,
                                                r1b[1], 1, 1, act2, 256);
  gemm_ws<<<dim3(2, ROWS/16), 128, 0, stream>>>(act2, 256, act2, 256, 256, r1w2s, 256, 128, 128,
                                                r1b[2], 1, 1, act3, 128);
  gemm_ws<<<dim3(1, ROWS/16), 128, 0, stream>>>(act3, 128, act3, 128, 128, r1w3s, 128, 16, 12,
                                                r1b[3], 0, 0, gatep, 16);
  gemm_ws<<<dim3(2, ROWS/16), 128, 0, stream>>>(hcur, HH, hcur, HH, HH, r2w0s, HH, 128, 128,
                                                r2b[0], 1, 1, act1, 128);
  gemm_ws<<<dim3(4, ROWS/16), 128, 0, stream>>>(act1, 128, act1, 128, 128, r2w1s, 128, 256, 256,
                                                r2b[1], 1, 1, act2, 256);
  gemm_ws<<<dim3(2, ROWS/16), 128, 0, stream>>>(act2, 256, act2, 256, 256, r2w2s, 256, 128, 128,
                                                r2b[2], 1, 1, act3, 128);
  gemm_ws<<<dim3(1, ROWS/16), 128, 0, stream>>>(act3, 128, act3, 128, 128, r2w3s, 128, 16, 12,
                                                r2b[3], 0, 0, valp, 16);

  readout_reduce<<<BB, 128, 0, stream>>>(gatep, valp, rmask, out);
}
